// DeepInfoMaxLoss_4458176053627
// MI455X (gfx1250) — matrix operand order repaired and checked by reading, not compiled, not run on hardware
//
#include <hip/hip_runtime.h>
#include <hip/hip_bf16.h>

// ---------------------------------------------------------------------------
// DeepInfoMax loss, MI455X (gfx1250), wave32 + WMMA bf16 pipeline.
// ---------------------------------------------------------------------------

typedef __bf16 bf16_t;
typedef __attribute__((ext_vector_type(8)))  __bf16 v8bf;
typedef __attribute__((ext_vector_type(16))) __bf16 v16bf;
typedef __attribute__((ext_vector_type(8)))  float  v8f;

#define B_SZ   8192
#define DIMD   1024
#define N_I    16
#define ALPHA_C 0.5f
#define BETA_C  1.0f

#if defined(__HIP_DEVICE_COMPILE__) && defined(__gfx1250__) &&                 \
    __has_builtin(__builtin_amdgcn_global_load_async_to_lds_b128) &&           \
    __has_builtin(__builtin_amdgcn_s_wait_asynccnt)
#define HAVE_ASYNC_LDS 1
#else
#define HAVE_ASYNC_LDS 0
#endif

#if HAVE_ASYNC_LDS
typedef int v4i_t __attribute__((vector_size(4 * sizeof(int))));
// async global->LDS copy of 16B per lane; tracked by ASYNCcnt
__device__ inline void async_cp_b128(const void* g, void* l) {
  __builtin_amdgcn_global_load_async_to_lds_b128(
      (__attribute__((address_space(1))) v4i_t*)g,
      (__attribute__((address_space(3))) v4i_t*)l, 0, 0);
}
#endif

__device__ inline v16bf cat16(v8bf lo, v8bf hi) {
  v16bf r;
#pragma unroll
  for (int i = 0; i < 8; ++i) { r[i] = lo[i]; r[i + 8] = hi[i]; }
  return r;
}

__device__ inline v8f wmma_bf16(v16bf a, v16bf b, v8f c) {
  return __builtin_amdgcn_wmma_f32_16x16x32_bf16(false, a, false, b, (short)0,
                                                 c, false, false);
}

__device__ inline float softplus_f(float x) {
  return fmaxf(x, 0.f) + log1pf(expf(-fabsf(x)));
}

// ------------------------------ conversions --------------------------------

__global__ void conv_f32_bf16_k(const float* __restrict__ s,
                                bf16_t* __restrict__ d, long n) {
  long i = (long)blockIdx.x * blockDim.x + threadIdx.x;
  long st = (long)gridDim.x * blockDim.x;
  for (; i < n; i += st) d[i] = (bf16_t)s[i];
}

// lW1 [16][1088][128] f32 -> Wm [16][64][128] bf16 (M-part)
//                            Wy [1024][16*128] bf16 (y-part, GEMM-friendly)
__global__ void split_lW1_k(const float* __restrict__ lW1,
                            bf16_t* __restrict__ Wy, bf16_t* __restrict__ Wm) {
  const long total = (long)N_I * 1088 * 128;
  long i = (long)blockIdx.x * blockDim.x + threadIdx.x;
  long st = (long)gridDim.x * blockDim.x;
  for (; i < total; i += st) {
    int h = (int)(i % 128);
    long rn = i / 128;
    int r = (int)(rn % 1088);
    int n = (int)(rn / 1088);
    bf16_t v = (bf16_t)lW1[i];
    if (r < 64) Wm[((long)n * 64 + r) * 128 + h] = v;
    else        Wy[(long)(r - 64) * (N_I * 128) + n * 128 + h] = v;
  }
}

__global__ void zero_k(float* p) { if (threadIdx.x == 0) *p = 0.f; }
__global__ void fin_k(const float* __restrict__ a, float* __restrict__ o) {
  if (threadIdx.x == 0) o[0] = *a;
}

// ------------------------- generic bf16 WMMA GEMM --------------------------
// C[M,N] = op(A[M,K] @ B[K,N] + bias). Tile 128x64, K-step 32, 8 waves,
// double-buffered LDS; A tile staged with async global->LDS when available.
// Requires M%128==0, N%64==0, K%32==0.

__global__ void __launch_bounds__(256)
gemm_bf16_k(const bf16_t* __restrict__ A, const bf16_t* __restrict__ Bm,
            const float* __restrict__ bias, float* __restrict__ Cf,
            bf16_t* __restrict__ Cb, int M, int N, int K, int relu) {
  __shared__ bf16_t As[2][128 * 40];  // row stride 40 bf16 (80B, 16B aligned)
  __shared__ bf16_t Bs[2][64 * 40];   // transposed: [col][k], stride 40
  (void)M;
  const int m0 = blockIdx.x * 128;
  const int n0 = blockIdx.y * 64;
  const int t = threadIdx.x;
  const int w = t >> 5, L = t & 31, lm = L & 15;

  // per-thread staging coordinates (two A chunks + one B chunk of 8 bf16)
  const int arow0 = t >> 2, acg = (t & 3) << 3;
  const int arow1 = arow0 + 64;
  const int bk = t >> 3, bcg = (t & 7) << 3;

  auto load_tile = [&](int k0, int buf) {
#if HAVE_ASYNC_LDS
    async_cp_b128(&A[(size_t)(m0 + arow0) * K + k0 + acg],
                  &As[buf][arow0 * 40 + acg]);
    async_cp_b128(&A[(size_t)(m0 + arow1) * K + k0 + acg],
                  &As[buf][arow1 * 40 + acg]);
#else
    *(v8bf*)&As[buf][arow0 * 40 + acg] =
        *(const v8bf*)&A[(size_t)(m0 + arow0) * K + k0 + acg];
    *(v8bf*)&As[buf][arow1 * 40 + acg] =
        *(const v8bf*)&A[(size_t)(m0 + arow1) * K + k0 + acg];
#endif
    v8bf bv = *(const v8bf*)&Bm[(size_t)(k0 + bk) * N + n0 + bcg];
#pragma unroll
    for (int i = 0; i < 8; ++i) Bs[buf][(bcg + i) * 40 + bk] = bv[i];
  };

  v8f acc[4] = {};
  const int nk = K >> 5;

  load_tile(0, 0);
#if HAVE_ASYNC_LDS
  __builtin_amdgcn_s_wait_asynccnt(0);
#endif
  __syncthreads();

  for (int ki = 0; ki < nk; ++ki) {
    const int buf = ki & 1;
    if (ki + 1 < nk) load_tile((ki + 1) << 5, buf ^ 1);
    if (ki + 2 < nk) {  // L2 prefetch two tiles ahead (global_prefetch_b8)
      __builtin_prefetch(&A[(size_t)(m0 + arow0) * K + ((ki + 2) << 5) + acg], 0, 0);
      __builtin_prefetch(&Bm[(size_t)(((ki + 2) << 5) + bk) * N + n0 + bcg], 0, 0);
    }

    const int r = w * 16 + lm;
    const int ab = r * 40 + ((L < 16) ? 0 : 8);
    v16bf af = cat16(*(v8bf*)&As[buf][ab], *(v8bf*)&As[buf][ab + 16]);
    v16bf bfr[4];
#pragma unroll
    for (int nt = 0; nt < 4; ++nt) {
      int c = nt * 16 + lm;
      int bb = c * 40 + ((L < 16) ? 0 : 8);
      bfr[nt] = cat16(*(v8bf*)&Bs[buf][bb], *(v8bf*)&Bs[buf][bb + 16]);
    }
#pragma unroll
    for (int nt = 0; nt < 4; ++nt) acc[nt] = wmma_bf16(af, bfr[nt], acc[nt]);

#if HAVE_ASYNC_LDS
    __builtin_amdgcn_s_wait_asynccnt(0);
#endif
    __syncthreads();
  }

#pragma unroll
  for (int nt = 0; nt < 4; ++nt) {
    int col = n0 + nt * 16 + lm;
    int rb = m0 + w * 16 + ((L < 16) ? 0 : 8);
    float bv = bias ? bias[col] : 0.f;
#pragma unroll
    for (int i = 0; i < 8; ++i) {
      float vv = acc[nt][i] + bv;
      if (relu) vv = fmaxf(vv, 0.f);
      if (Cf) Cf[(size_t)(rb + i) * N + col] = vv;
      if (Cb) Cb[(size_t)(rb + i) * N + col] = (bf16_t)vv;
    }
  }
}

// -------------------- mpart grouped GEMM  [B,64]x[64,128] ------------------
// mpart[b, n*128+h] = sum_j Mb[b, j*16+n] * Wm[n][j][h]

__global__ void __launch_bounds__(256)
mpart_k(const bf16_t* __restrict__ Mb, const bf16_t* __restrict__ Wm,
        bf16_t* __restrict__ outp) {
  __shared__ bf16_t As[128 * 72];  // stride 72 bf16 (144B)
  __shared__ bf16_t Bs[128 * 72];  // [col][k]
  const int bm = blockIdx.x * 128;
  const int n = blockIdx.y;
  const int t = threadIdx.x, w = t >> 5, L = t & 31, lm = L & 15;

  {  // strided gather of A (column n of each group-of-16)
    int row = t >> 1, jb = (t & 1) * 32;
    const bf16_t* src = Mb + (size_t)(bm + row) * DIMD + n;
    for (int j = 0; j < 32; ++j)
      As[row * 72 + jb + j] = src[(jb + j) * N_I];
  }
  {  // Wm[n] [64][128] -> transposed
    int k = t >> 2, cb = (t & 3) * 32;
    const bf16_t* src = Wm + ((size_t)n * 64 + k) * 128 + cb;
    for (int i = 0; i < 32; ++i) Bs[(cb + i) * 72 + k] = src[i];
  }
  __syncthreads();

  v8f acc[8] = {};
#pragma unroll
  for (int ks = 0; ks < 2; ++ks) {
    int k0 = ks * 32;
    int r = w * 16 + lm;
    int ab = r * 72 + k0 + ((L < 16) ? 0 : 8);
    v16bf af = cat16(*(v8bf*)&As[ab], *(v8bf*)&As[ab + 16]);
#pragma unroll
    for (int nt = 0; nt < 8; ++nt) {
      int c = nt * 16 + lm;
      int bb = c * 72 + k0 + ((L < 16) ? 0 : 8);
      v16bf bf = cat16(*(v8bf*)&Bs[bb], *(v8bf*)&Bs[bb + 16]);
      acc[nt] = wmma_bf16(af, bf, acc[nt]);
    }
  }

#pragma unroll
  for (int nt = 0; nt < 8; ++nt) {
    int col = nt * 16 + lm;
    int rb = bm + w * 16 + ((L < 16) ? 0 : 8);
#pragma unroll
    for (int i = 0; i < 8; ++i)
      outp[(size_t)(rb + i) * (N_I * 128) + n * 128 + col] = (bf16_t)acc[nt][i];
  }
}

// ---------------- fused local head: relu-add -> WMMA -> dot -> sp ----------
// grid (B/64, 16, 2[pos/neg]), 4 waves. LDS 52KB.

__global__ void __launch_bounds__(128)
local_head_k(const bf16_t* __restrict__ ypart, const bf16_t* __restrict__ mpart,
             const float* __restrict__ lb1, const bf16_t* __restrict__ lW2b,
             const float* __restrict__ lb2, const float* __restrict__ lW3,
             const float* __restrict__ lb3, float* __restrict__ accp) {
  __shared__ bf16_t As[64 * 136];
  __shared__ bf16_t Bs[128 * 136];
  const int bm = blockIdx.x * 64;
  const int n = blockIdx.y;
  const int sign = blockIdx.z;  // 0 = positive (yM), 1 = negative (yMp)
  const int t = threadIdx.x, w = t >> 5, L = t & 31, lm = L & 15;

  {  // h1 = relu(ypart + mpart(rolled for neg) + lb1) -> bf16
    int row = t >> 1, hb = (t & 1) * 64;
    int grow = bm + row;
    int mrow = sign ? ((grow + 1) & (B_SZ - 1)) : grow;
    const bf16_t* yp = ypart + (size_t)grow * (N_I * 128) + n * 128 + hb;
    const bf16_t* mp = mpart + (size_t)mrow * (N_I * 128) + n * 128 + hb;
    const float* bb = lb1 + n * 128 + hb;
    for (int h = 0; h < 64; ++h) {
      float vv = (float)yp[h] + (float)mp[h] + bb[h];
      As[row * 136 + hb + h] = (bf16_t)fmaxf(vv, 0.f);
    }
  }
  {  // lW2[n] [128][128] -> transposed
    const bf16_t* src = lW2b + (size_t)n * 16384 + (size_t)t * 128;
    for (int i = 0; i < 128; ++i) Bs[i * 136 + t] = src[i];
  }
  __syncthreads();

  v8f acc[8] = {};
#pragma unroll
  for (int ks = 0; ks < 4; ++ks) {
    int k0 = ks * 32;
    int r = w * 16 + lm;
    int ab = r * 136 + k0 + ((L < 16) ? 0 : 8);
    v16bf af = cat16(*(v8bf*)&As[ab], *(v8bf*)&As[ab + 16]);
#pragma unroll
    for (int nt = 0; nt < 8; ++nt) {
      int c = nt * 16 + lm;
      int bb = c * 136 + k0 + ((L < 16) ? 0 : 8);
      v16bf bf = cat16(*(v8bf*)&Bs[bb], *(v8bf*)&Bs[bb + 16]);
      acc[nt] = wmma_bf16(af, bf, acc[nt]);
    }
  }

  // h2 = relu(acc + lb2); s = h2 . lW3[n] + lb3[n]; softplus; accumulate.
  float part[8];
#pragma unroll
  for (int i = 0; i < 8; ++i) part[i] = 0.f;
#pragma unroll
  for (int nt = 0; nt < 8; ++nt) {
    float b2 = lb2[n * 128 + nt * 16 + lm];
    float w3 = lW3[n * 128 + nt * 16 + lm];
#pragma unroll
    for (int i = 0; i < 8; ++i)
      part[i] += fmaxf(acc[nt][i] + b2, 0.f) * w3;
  }
#pragma unroll
  for (int m = 1; m < 16; m <<= 1)
#pragma unroll
    for (int i = 0; i < 8; ++i) part[i] += __shfl_xor(part[i], m, 32);

  if (lm == 0) {  // lanes 0 (rows 0-7) and 16 (rows 8-15)
    float b3 = lb3[n];
    float local = 0.f;
#pragma unroll
    for (int i = 0; i < 8; ++i) {
      float s = part[i] + b3;
      local += sign ? softplus_f(s) : softplus_f(-s);
    }
    atomicAdd(accp, local * (BETA_C / (float)(B_SZ * N_I)));
  }
}

// ---------------- fused global head: relu-add -> WMMA -> dot -> sp ---------
// grid (B/64, 2[pos/neg]), 4 waves.

__global__ void __launch_bounds__(128)
global_head_k(const float* __restrict__ u, const float* __restrict__ v,
              const float* __restrict__ l0b, const bf16_t* __restrict__ l1wb,
              const float* __restrict__ l1b, const float* __restrict__ l2w,
              const float* __restrict__ l2b, float* __restrict__ accp) {
  __shared__ bf16_t As[64 * 136];
  __shared__ bf16_t Bs[128 * 136];
  const int bm = blockIdx.x * 64;
  const int sign = blockIdx.y;
  const int t = threadIdx.x, w = t >> 5, L = t & 31, lm = L & 15;

  {  // h0 = relu(u + v(rolled for neg) + l0b)
    int row = t >> 1, hb = (t & 1) * 64;
    int grow = bm + row;
    int vrow = sign ? ((grow + 1) & (B_SZ - 1)) : grow;
    const float* up = u + (size_t)grow * 128 + hb;
    const float* vp = v + (size_t)vrow * 128 + hb;
    const float* bb = l0b + hb;
    for (int h = 0; h < 64; ++h)
      As[row * 136 + hb + h] = (bf16_t)fmaxf(up[h] + vp[h] + bb[h], 0.f);
  }
  {  // l1w [128][128] -> transposed
    const bf16_t* src = l1wb + (size_t)t * 128;
    for (int i = 0; i < 128; ++i) Bs[i * 136 + t] = src[i];
  }
  __syncthreads();

  v8f acc[8] = {};
#pragma unroll
  for (int ks = 0; ks < 4; ++ks) {
    int k0 = ks * 32;
    int r = w * 16 + lm;
    int ab = r * 136 + k0 + ((L < 16) ? 0 : 8);
    v16bf af = cat16(*(v8bf*)&As[ab], *(v8bf*)&As[ab + 16]);
#pragma unroll
    for (int nt = 0; nt < 8; ++nt) {
      int c = nt * 16 + lm;
      int bb = c * 136 + k0 + ((L < 16) ? 0 : 8);
      v16bf bf = cat16(*(v8bf*)&Bs[bb], *(v8bf*)&Bs[bb + 16]);
      acc[nt] = wmma_bf16(af, bf, acc[nt]);
    }
  }

  float part[8];
#pragma unroll
  for (int i = 0; i < 8; ++i) part[i] = 0.f;
#pragma unroll
  for (int nt = 0; nt < 8; ++nt) {
    float b1 = l1b[nt * 16 + lm];
    float wv = l2w[nt * 16 + lm];
#pragma unroll
    for (int i = 0; i < 8; ++i)
      part[i] += fmaxf(acc[nt][i] + b1, 0.f) * wv;
  }
#pragma unroll
  for (int m = 1; m < 16; m <<= 1)
#pragma unroll
    for (int i = 0; i < 8; ++i) part[i] += __shfl_xor(part[i], m, 32);

  if (lm == 0) {
    float b2 = l2b[0];
    float local = 0.f;
#pragma unroll
    for (int i = 0; i < 8; ++i) {
      float s = part[i] + b2;
      local += sign ? softplus_f(s) : softplus_f(-s);
    }
    atomicAdd(accp, local * (ALPHA_C / (float)B_SZ));
  }
}

// ------------------------------- launcher ----------------------------------

extern "C" void kernel_launch(void* const* d_in, const int* in_sizes, int n_in,
                              void* d_out, int out_size, void* d_ws,
                              size_t ws_size, hipStream_t stream) {
  (void)in_sizes; (void)n_in; (void)out_size; (void)ws_size;
  const float* y   = (const float*)d_in[0];
  const float* M   = (const float*)d_in[1];
  const float* gw0 = (const float*)d_in[2];
  const float* gb0 = (const float*)d_in[3];
  const float* gw1 = (const float*)d_in[4];
  const float* gb1 = (const float*)d_in[5];
  const float* l0w = (const float*)d_in[6];
  const float* l0b = (const float*)d_in[7];
  const float* l1w = (const float*)d_in[8];
  const float* l1b = (const float*)d_in[9];
  const float* l2w = (const float*)d_in[10];
  const float* l2b = (const float*)d_in[11];
  const float* lW1 = (const float*)d_in[12];
  const float* lb1 = (const float*)d_in[13];
  const float* lW2 = (const float*)d_in[14];
  const float* lb2 = (const float*)d_in[15];
  const float* lW3 = (const float*)d_in[16];
  const float* lb3 = (const float*)d_in[17];

  char* ws = (char*)d_ws;
  size_t off = 0;
  auto alloc = [&](size_t bytes) {
    size_t o = off;
    off = (off + bytes + 255) & ~(size_t)255;
    return o;
  };

  float*  acc   = (float*)(ws + alloc(4));
  bf16_t* yb    = (bf16_t*)(ws + alloc((size_t)B_SZ * DIMD * 2));
  bf16_t* Mb    = (bf16_t*)(ws + alloc((size_t)B_SZ * DIMD * 2));
  bf16_t* gw0b  = (bf16_t*)(ws + alloc((size_t)DIMD * DIMD * 2));
  bf16_t* gw1b  = (bf16_t*)(ws + alloc((size_t)DIMD * DIMD * 2));
  bf16_t* l0wb  = (bf16_t*)(ws + alloc((size_t)2 * DIMD * 128 * 2));
  bf16_t* l1wb  = (bf16_t*)(ws + alloc((size_t)128 * 128 * 2));
  bf16_t* Wy    = (bf16_t*)(ws + alloc((size_t)DIMD * N_I * 128 * 2));
  bf16_t* Wm    = (bf16_t*)(ws + alloc((size_t)N_I * 64 * 128 * 2));
  bf16_t* lW2b  = (bf16_t*)(ws + alloc((size_t)N_I * 128 * 128 * 2));
  float*  u     = (float*)(ws + alloc((size_t)B_SZ * 128 * 4));
  float*  v     = (float*)(ws + alloc((size_t)B_SZ * 128 * 4));
  bf16_t* ypart = (bf16_t*)(ws + alloc((size_t)B_SZ * N_I * 128 * 2));
  bf16_t* mpart = (bf16_t*)(ws + alloc((size_t)B_SZ * N_I * 128 * 2));
  bf16_t* g1b = ypart;  // dead before ypart is written (stream-ordered)
  bf16_t* gb  = mpart;  // dead before mpart is written (stream-ordered)

  zero_k<<<1, 32, 0, stream>>>(acc);

  conv_f32_bf16_k<<<4096, 256, 0, stream>>>(y, yb, (long)B_SZ * DIMD);
  conv_f32_bf16_k<<<4096, 256, 0, stream>>>(M, Mb, (long)B_SZ * DIMD);
  conv_f32_bf16_k<<<1024, 256, 0, stream>>>(gw0, gw0b, (long)DIMD * DIMD);
  conv_f32_bf16_k<<<1024, 256, 0, stream>>>(gw1, gw1b, (long)DIMD * DIMD);
  conv_f32_bf16_k<<<256, 256, 0, stream>>>(l0w, l0wb, (long)2 * DIMD * 128);
  conv_f32_bf16_k<<<64, 256, 0, stream>>>(l1w, l1wb, (long)128 * 128);
  conv_f32_bf16_k<<<256, 256, 0, stream>>>(lW2, lW2b, (long)N_I * 128 * 128);
  split_lW1_k<<<2048, 256, 0, stream>>>(lW1, Wy, Wm);

  // g1 = relu(M @ gw0 + gb0)                    [8192,1024]
  gemm_bf16_k<<<dim3(B_SZ / 128, DIMD / 64), 256, 0, stream>>>(
      Mb, gw0b, gb0, nullptr, g1b, B_SZ, DIMD, DIMD, 1);
  // g  = g1 @ gw1 + gb1                         [8192,1024]
  gemm_bf16_k<<<dim3(B_SZ / 128, DIMD / 64), 256, 0, stream>>>(
      g1b, gw1b, gb1, nullptr, gb, B_SZ, DIMD, DIMD, 0);
  // u = y @ l0w[:1024]                          [8192,128]
  gemm_bf16_k<<<dim3(B_SZ / 128, 2), 256, 0, stream>>>(
      yb, l0wb, nullptr, u, nullptr, B_SZ, 128, DIMD, 0);
  // v = g @ l0w[1024:]                          [8192,128]
  gemm_bf16_k<<<dim3(B_SZ / 128, 2), 256, 0, stream>>>(
      gb, l0wb + (size_t)DIMD * 128, nullptr, v, nullptr, B_SZ, 128, DIMD, 0);
  // ypart = y @ Wy                              [8192,2048]  (overwrites g1b)
  gemm_bf16_k<<<dim3(B_SZ / 128, (N_I * 128) / 64), 256, 0, stream>>>(
      yb, Wy, nullptr, nullptr, ypart, B_SZ, N_I * 128, DIMD, 0);
  // mpart grouped GEMMs                          (overwrites gb)
  mpart_k<<<dim3(B_SZ / 128, N_I), 256, 0, stream>>>(Mb, Wm, mpart);

  // fused heads (pos + neg), accumulate into acc
  local_head_k<<<dim3(B_SZ / 64, N_I, 2), 128, 0, stream>>>(
      ypart, mpart, lb1, lW2b, lb2, lW3, lb3, acc);
  global_head_k<<<dim3(B_SZ / 64, 2), 128, 0, stream>>>(
      u, v, l0b, l1wb, l1b, l2w, l2b, acc);

  fin_k<<<1, 32, 0, stream>>>(acc, (float*)d_out);
}
